// SubNetwork_43903155700175
// MI455X (gfx1250) — matrix operand order repaired and checked by reading, not compiled, not run on hardware
//
#include <hip/hip_runtime.h>
#include <math.h>

// ---------------------------------------------------------------------------
// GNN-style 3-layer: (Linear -> LayerNorm -> ReLU) -> gather(src) -> segmax(16)
// N=50000 nodes, H=64, layer0 K=64, layers1/2 K=128 (concat input).
// Memory-bound workload (~2 GFLOP, ~120MB HBM traffic); gathers are L2-resident
// (each 25.6MB cat buffer << 192MB L2). GEMM runs on v_wmma_f32_16x16x32_f16.
// ---------------------------------------------------------------------------

typedef _Float16 v16h __attribute__((ext_vector_type(16)));
typedef float    v8f  __attribute__((ext_vector_type(8)));

#define N_NODES 50000
#define DEG     16
#define HID     64
#define LN_EPS  1e-5f
#define N_TILES ((N_NODES + 15) / 16)   // 3125 exactly

// One wave32 computes a 16-node x 64-feature tile:
//   Y[16xH] = relu( LN( X[16xK] @ W^T[KxH] + bias ) )
// and stores it into the first half of the 2H-wide concat buffer (ldy=128).
template<int K>
__global__ __launch_bounds__(128)
void gemm_ln_relu_kernel(const float* __restrict__ X,   int ldx,
                         const float* __restrict__ W,           // [H, K] row-major
                         const float* __restrict__ bias,
                         const float* __restrict__ gamma,
                         const float* __restrict__ beta,
                         float* __restrict__ Y,         int ldy)
{
    const int wave = threadIdx.x >> 5;
    const int lane = threadIdx.x & 31;
    const int tile = blockIdx.x * 4 + wave;
    if (tile >= N_TILES) return;               // wave-uniform: EXEC stays all-1s

    const int node0 = tile * 16;
    const int m     = lane & 15;               // row within tile / column id
    const int half  = lane >> 4;               // lane-group select

    v8f acc[4];
    #pragma unroll
    for (int t = 0; t < 4; ++t) acc[t] = v8f{};

    const float* xrow = X + (size_t)(node0 + m) * ldx;

    #pragma unroll
    for (int kc = 0; kc < K / 32; ++kc) {
        // ---- A fragment: 16x32 f16, ISA layout:
        // lane m holds row m; VGPR j holds K = 16*(j/4) + 8*half + 2*(j&3)
        v16h a;
        #pragma unroll
        for (int j = 0; j < 8; ++j) {
            const int k = kc * 32 + (j >> 2) * 16 + half * 8 + 2 * (j & 3);
            const float2 p = *(const float2*)(xrow + k);
            a[2 * j]     = (_Float16)p.x;
            a[2 * j + 1] = (_Float16)p.y;
        }
        // ---- 4 B fragments (one per 16-wide N tile): B[k][n] = W[n][k]
        // lane holds column n = 16*t + m; VGPR j holds K = 16*(j/4) + 8*half + 2*(j&3)
        #pragma unroll
        for (int t = 0; t < 4; ++t) {
            const float* wrow = W + (size_t)(t * 16 + m) * K;
            v16h b;
            #pragma unroll
            for (int j = 0; j < 8; ++j) {
                const int k = kc * 32 + (j >> 2) * 16 + half * 8 + 2 * (j & 3);
                const float2 p = *(const float2*)(wrow + k);
                b[2 * j]     = (_Float16)p.x;
                b[2 * j + 1] = (_Float16)p.y;
            }
            acc[t] = __builtin_amdgcn_wmma_f32_16x16x32_f16(
                         false, a, false, b, (short)0, acc[t], false, false);
        }
    }

    // ---- bias (n = 16*t + m, identical for all 8 vgprs of a tile's acc)
    float bia[4], gam[4], bet[4];
    #pragma unroll
    for (int t = 0; t < 4; ++t) {
        const int n = t * 16 + m;
        bia[t] = bias[n]; gam[t] = gamma[n]; bet[t] = beta[n];
    }
    #pragma unroll
    for (int t = 0; t < 4; ++t)
        #pragma unroll
        for (int i = 0; i < 8; ++i) acc[t][i] += bia[t];

    // ---- LayerNorm + ReLU + store.
    // D layout: element (mrow, n): vgpr r = mrow%8, lane = (mrow/8)*16 + n%16.
    // Row (half*8 + r) lives at vgpr index r across the 16 lanes of this half.
    #pragma unroll
    for (int r = 0; r < 8; ++r) {
        const float v0 = acc[0][r], v1 = acc[1][r], v2 = acc[2][r], v3 = acc[3][r];
        float s = v0 + v1 + v2 + v3;
        float q = v0 * v0 + v1 * v1 + v2 * v2 + v3 * v3;
        #pragma unroll
        for (int off = 1; off < 16; off <<= 1) {   // stays within 16-lane half
            s += __shfl_xor(s, off, 32);
            q += __shfl_xor(q, off, 32);
        }
        const float mu   = s * (1.0f / 64.0f);
        const float var  = q * (1.0f / 64.0f) - mu * mu;
        const float rstd = rsqrtf(var + LN_EPS);

        const size_t row = (size_t)(node0 + half * 8 + r) * ldy;
        const float vv[4] = {v0, v1, v2, v3};
        #pragma unroll
        for (int t = 0; t < 4; ++t) {
            float y = (vv[t] - mu) * rstd * gam[t] + bet[t];
            y = fmaxf(y, 0.0f);
            Y[row + t * 16 + m] = y;   // 16 consecutive lanes -> 64B coalesced
        }
    }
}

// Per-node max over its DEG=16 edges (dst = repeat(arange(N), DEG), sorted),
// reading h rows (first half of cat, stride 128) from L2, writing second half.
__global__ __launch_bounds__(256)
void seg_max_kernel(const float* __restrict__ cat,
                    const int* __restrict__ src,
                    float* __restrict__ out)
{
    const long long idx = (long long)blockIdx.x * blockDim.x + threadIdx.x;
    const int node = (int)(idx >> 6);
    const int n    = (int)(idx & 63);
    if (node >= N_NODES) return;

    const int* sp = src + (size_t)node * DEG;
    float m = -INFINITY;
    #pragma unroll
    for (int e = 0; e < DEG; ++e) {
        const int s = sp[e];                       // uniform per 64-thread group
        m = fmaxf(m, cat[(size_t)s * 128 + n]);    // 256B coalesced gather (L2)
    }
    out[(size_t)node * 128 + 64 + n] = m;
}

extern "C" void kernel_launch(void* const* d_in, const int* in_sizes, int n_in,
                              void* d_out, int out_size, void* d_ws, size_t ws_size,
                              hipStream_t stream)
{
    const float* x   = (const float*)d_in[0];
    const int*   src = (const int*)  d_in[1];
    // d_in[2] = dst (implicit: repeat(arange(N), DEG)) -- unused
    const float* W0  = (const float*)d_in[3];
    const float* b0  = (const float*)d_in[4];
    const float* g0  = (const float*)d_in[5];
    const float* be0 = (const float*)d_in[6];
    const float* W1  = (const float*)d_in[7];
    const float* b1  = (const float*)d_in[8];
    const float* g1  = (const float*)d_in[9];
    const float* be1 = (const float*)d_in[10];
    const float* W2  = (const float*)d_in[11];
    const float* b2  = (const float*)d_in[12];
    const float* g2  = (const float*)d_in[13];
    const float* be2 = (const float*)d_in[14];

    float* cat0 = (float*)d_ws;                          // [N, 128] = 25.6 MB
    float* cat1 = cat0 + (size_t)N_NODES * 128;          // [N, 128] = 25.6 MB
    float* outp = (float*)d_out;                         // [N, 128]

    const int ggrid = (N_TILES + 3) / 4;                       // 4 waves / block
    const int agrid = ((long long)N_NODES * 64 + 255) / 256;

    // Layer 0: K = 64 (reads x), writes cat0[:, 0:64] then cat0[:, 64:128]
    gemm_ln_relu_kernel<64><<<ggrid, 128, 0, stream>>>(x, 64, W0, b0, g0, be0, cat0, 128);
    seg_max_kernel<<<agrid, 256, 0, stream>>>(cat0, src, cat0);

    // Layer 1: K = 128 (reads cat0), writes cat1
    gemm_ln_relu_kernel<128><<<ggrid, 128, 0, stream>>>(cat0, 128, W1, b1, g1, be1, cat1, 128);
    seg_max_kernel<<<agrid, 256, 0, stream>>>(cat1, src, cat1);

    // Layer 2: K = 128 (reads cat1), writes straight into d_out
    gemm_ln_relu_kernel<128><<<ggrid, 128, 0, stream>>>(cat1, 128, W2, b2, g2, be2, outp, 128);
    seg_max_kernel<<<agrid, 256, 0, stream>>>(outp, src, outp);
}
